// CosineAttention_21397527068866
// MI455X (gfx1250) — compile-verified, matrix operand order checked by reading, not run on hardware
//
#include <hip/hip_runtime.h>
#include <hip/hip_bf16.h>

// ---------------------------------------------------------------------------
// CosineAttention fused kernel for MI455X (gfx1250, wave32, WMMA).
//
//   out = sigmoid( (X X^T / (||x_i|| ||x_j||)) @ X ),  X: [8192,1024] f32
//
// bf16 WMMA (16x16x32) for both GEMMs, flash-style fusion (no 256MB sim
// matrix), async-DMA staging of Xj tiles into a double-buffered LDS region
// (GLOBAL_LOAD_ASYNC_TO_LDS_B128 / ASYNCcnt) pipelined against the WMMA
// K-loop, scale phase and the second GEMM.
// ---------------------------------------------------------------------------

typedef __bf16        bf16_t;
typedef bf16_t        v8bf   __attribute__((ext_vector_type(8)));
typedef bf16_t        v16bf  __attribute__((ext_vector_type(16)));
typedef float         v8f    __attribute__((ext_vector_type(8)));
typedef float         v4f    __attribute__((ext_vector_type(4)));
typedef unsigned short v4us  __attribute__((ext_vector_type(4)));

#define N_ROWS 8192
#define D_COLS 1024

#define BJ       32            // j-chunk (== K of GEMM2)
#define KHALF    512           // stage half of the D=1024 K range at a time
#define XJ_PITCH 520           // 512 + 8 pad: 1040B row stride -> 4-bank skew, 16B aligned
#define S_PITCH  56            // 32 + 24 pad: 112B row stride -> conflict-free, 16B aligned

// f32 -> bf16 round-to-nearest-even (NaNs don't occur in this workload)
__device__ __forceinline__ unsigned short f32_to_bf16(float f) {
  unsigned u = __builtin_bit_cast(unsigned, f);
  u += 0x7FFFu + ((u >> 16) & 1u);
  return (unsigned short)(u >> 16);
}

// Build a 16-element bf16 operand from two contiguous 8-element (16B) chunks.
// Elements 0..7 -> VGPRs 0..3 (K octet at kbase), 8..15 -> VGPRs 4..7 (kbase+16),
// matching the CDNA5 16-bit A/B operand layout (two K-octets per lane).
__device__ __forceinline__ v16bf load_ab(const unsigned short* lo,
                                         const unsigned short* hi) {
  v8bf a = *reinterpret_cast<const v8bf*>(lo);
  v8bf b = *reinterpret_cast<const v8bf*>(hi);
  v16bf r;
#pragma unroll
  for (int t = 0; t < 8; ++t) { r[t] = a[t]; r[t + 8] = b[t]; }
  return r;
}

// Async DMA one 32x512 bf16 stage of Xj into an LDS buffer. Fire-and-forget:
// tracked by ASYNCcnt, no VGPR landing, completion = s_wait_asynccnt + barrier.
// VDST operand is the wave-relative LDS byte offset (low 32 bits of the
// generic shared-aperture address per the CDNA5 flat-addressing rules).
__device__ __forceinline__ void stage_async(const unsigned short* __restrict__ xb,
                                            int j0, int half,
                                            unsigned short* xjbuf, int tid) {
  const unsigned short* src = xb + (size_t)j0 * D_COLS + half * KHALF;
#pragma unroll
  for (int it = 0; it < 8; ++it) {
    int q  = it * 256 + tid;            // 2048 16B chunks over 256 threads
    int jr = q >> 6;                    // 64 chunks per 512-elem row
    int jc = (q & 63) * 8;
    unsigned ldsoff =
        (unsigned)(unsigned long long)(&xjbuf[jr * XJ_PITCH + jc]);
    unsigned long long ga =
        (unsigned long long)(src + (size_t)jr * D_COLS + jc);
    asm volatile("global_load_async_to_lds_b128 %0, %1, off"
                 :: "v"(ldsoff), "v"(ga) : "memory");
  }
}

__device__ __forceinline__ void wait_async_all() {
  asm volatile("s_wait_asynccnt 0x0" ::: "memory");
}

// ---------------------------------------------------------------------------
// Prep 1: per-row inverse norm + bf16 copy of X (coalesced).
// ---------------------------------------------------------------------------
__global__ __launch_bounds__(256) void prep_norms_kernel(
    const float* __restrict__ x, unsigned short* __restrict__ xb,
    float* __restrict__ inv_n) {
  __shared__ float red[256];
  const int row = blockIdx.x;
  const int t   = threadIdx.x;
  const float* xr = x + (size_t)row * D_COLS;
  v4f v = reinterpret_cast<const v4f*>(xr)[t];
  float ss = v[0] * v[0] + v[1] * v[1] + v[2] * v[2] + v[3] * v[3];
  v4us o;
#pragma unroll
  for (int e = 0; e < 4; ++e) o[e] = f32_to_bf16(v[e]);
  reinterpret_cast<v4us*>(xb + (size_t)row * D_COLS)[t] = o;
  red[t] = ss;
  __syncthreads();
#pragma unroll
  for (int s = 128; s > 0; s >>= 1) {
    if (t < s) red[t] += red[t + s];
    __syncthreads();
  }
  // ||x|| ~ 32 for N(0,1) rows with D=1024; eps=1e-8 clamp never binds.
  if (t == 0) inv_n[row] = rsqrtf(red[0] + 1e-30f);
}

// ---------------------------------------------------------------------------
// Prep 2: X^T in bf16 (LDS-tiled 64x64 transpose, coalesced both sides).
// ---------------------------------------------------------------------------
__global__ __launch_bounds__(256) void transpose_bf16_kernel(
    const float* __restrict__ x, unsigned short* __restrict__ xbt) {
  __shared__ unsigned short tile[64][72];  // +8 pad to skew banks
  const int r0 = blockIdx.x * 64;
  const int c0 = blockIdx.y * 64;
  const int t  = threadIdx.x;
#pragma unroll
  for (int p = 0; p < 16; ++p) {
    int idx = p * 256 + t;
    int r = idx >> 6, c = idx & 63;
    tile[r][c] = f32_to_bf16(x[(size_t)(r0 + r) * D_COLS + (c0 + c)]);
  }
  __syncthreads();
#pragma unroll
  for (int p = 0; p < 16; ++p) {
    int idx = p * 256 + t;
    int cc = idx >> 6, rr = idx & 63;
    xbt[(size_t)(c0 + cc) * N_ROWS + (r0 + rr)] = tile[rr][cc];
  }
}

// ---------------------------------------------------------------------------
// Main fused kernel. Grid (64, 8); block 256 threads (8 wave32s).
// Phase 1 (sim): wave w owns S rows 16w..16w+15, accumulates over K=1024 with
//   A streamed from L2 and B read from the async-staged LDS Xj buffer.
// Phase 4 (out): waves form a 4x2 grid of 32x64 sub-tiles; A = S from LDS,
//   B = columns of X read from precomputed bf16 X^T. 64 f32 acc VGPRs/lane.
// ---------------------------------------------------------------------------
__global__ __launch_bounds__(256) void cosattn_main_kernel(
    const unsigned short* __restrict__ xb,   // [N,D] bf16
    const unsigned short* __restrict__ xbt,  // [D,N] bf16
    const float* __restrict__ inv_n,         // [N]
    float* __restrict__ out) {               // [N,D] f32
  __shared__ unsigned short XJ[2][BJ * XJ_PITCH];  // 2 x 33280 B (double buffer)
  __shared__ unsigned short SL[128 * S_PITCH];     // 14336 B: bf16 sim tile

  const int i0   = blockIdx.x * 128;
  const int d0   = blockIdx.y * 128;
  const int tid  = threadIdx.x;
  const int w    = tid >> 5;
  const int lane = tid & 31;
  const int lmod = lane & 15;
  const int lhi  = lane >> 4;
  const int wr   = w >> 1, wc = w & 1;

  // inverse norms for this wave's S rows (C-layout: VGPR r -> row 8*lhi + r)
  float invi[8];
#pragma unroll
  for (int r = 0; r < 8; ++r) invi[r] = inv_n[i0 + 16 * w + 8 * lhi + r];

  v8f acc[2][4];
#pragma unroll
  for (int a = 0; a < 2; ++a)
#pragma unroll
    for (int b = 0; b < 4; ++b)
#pragma unroll
      for (int e = 0; e < 8; ++e) acc[a][b][e] = 0.0f;

  // A-operand row for phase 1: lane lmod -> row i0 + 16w + lmod
  const unsigned short* arow = xb + (size_t)(i0 + 16 * w + lmod) * D_COLS;

  // Prologue of the async pipeline: kick off stage (j0=0, half=0).
  stage_async(xb, 0, 0, XJ[0], tid);

  for (int j0 = 0; j0 < N_ROWS; j0 += BJ) {
    v8f s0, s1;
#pragma unroll
    for (int e = 0; e < 8; ++e) { s0[e] = 0.0f; s1[e] = 0.0f; }

    // ---- Phase 1: S = Xi . Xj^T over K=1024, two async-staged halves -----
#pragma unroll 1
    for (int half = 0; half < 2; ++half) {
      // Entry: drain own async loads, then converge. After this barrier the
      // current buffer is fully written and the *other* buffer is idle.
      wait_async_all();
      __syncthreads();

      // Issue the next stage's DMA into the other buffer; it overlaps the
      // K-loop below (and, for half==1, also the scale phase and GEMM2).
      int nj0 = j0, nhalf = half + 1;
      if (nhalf == 2) { nj0 += BJ; nhalf = 0; }
      if (nj0 < N_ROWS) stage_async(xb, nj0, nhalf, XJ[nhalf], tid);

      const unsigned short* xjb = XJ[half];
#pragma unroll 4
      for (int k = 0; k < KHALF; k += 32) {
        const int kg = half * KHALF + k;
        v16bf a  = load_ab(arow + kg + lhi * 8, arow + kg + 16 + lhi * 8);
        v16bf b0 = load_ab(&xjb[lmod * XJ_PITCH + k + lhi * 8],
                           &xjb[lmod * XJ_PITCH + k + 16 + lhi * 8]);
        v16bf b1 = load_ab(&xjb[(16 + lmod) * XJ_PITCH + k + lhi * 8],
                           &xjb[(16 + lmod) * XJ_PITCH + k + 16 + lhi * 8]);
        s0 = __builtin_amdgcn_wmma_f32_16x16x32_bf16(false, a, false, b0,
                                                     (short)0, s0, false, false);
        s1 = __builtin_amdgcn_wmma_f32_16x16x32_bf16(false, a, false, b1,
                                                     (short)0, s1, false, false);
      }
    }

    // ---- Phase 2/3: scale by 1/(||xi|| ||xj||), bf16, relayout into LDS --
    const float ivj0 = inv_n[j0 + lmod];
    const float ivj1 = inv_n[j0 + 16 + lmod];
#pragma unroll
    for (int r = 0; r < 8; ++r) {
      int srow = 16 * w + 8 * lhi + r;
      SL[srow * S_PITCH + lmod]      = f32_to_bf16(s0[r] * invi[r] * ivj0);
      SL[srow * S_PITCH + 16 + lmod] = f32_to_bf16(s1[r] * invi[r] * ivj1);
    }
    __syncthreads();

    // ---- Phase 4: OUT += S[128x32] @ Xj_cols[32x128] ---------------------
    v16bf At[2], Bt[4];
#pragma unroll
    for (int rt = 0; rt < 2; ++rt) {
      int srow = 32 * wr + 16 * rt + lmod;
      At[rt] = load_ab(&SL[srow * S_PITCH + lhi * 8],
                       &SL[srow * S_PITCH + 16 + lhi * 8]);
    }
#pragma unroll
    for (int ct = 0; ct < 4; ++ct) {
      const unsigned short* bcol =
          xbt + (size_t)(d0 + 64 * wc + 16 * ct + lmod) * N_ROWS + j0;
      Bt[ct] = load_ab(bcol + lhi * 8, bcol + 16 + lhi * 8);
    }
#pragma unroll
    for (int rt = 0; rt < 2; ++rt)
#pragma unroll
      for (int ct = 0; ct < 4; ++ct)
        acc[rt][ct] = __builtin_amdgcn_wmma_f32_16x16x32_bf16(
            false, At[rt], false, Bt[ct], (short)0, acc[rt][ct], false, false);
    // No trailing barrier: the next stage's entry barrier (wait_async +
    // __syncthreads) already orders these SL reads before the next SL writes.
  }

  // ---- Epilogue: sigmoid + coalesced f32 stores --------------------------
#pragma unroll
  for (int rt = 0; rt < 2; ++rt)
#pragma unroll
    for (int ct = 0; ct < 4; ++ct)
#pragma unroll
      for (int r = 0; r < 8; ++r) {
        int row = i0 + 32 * wr + 16 * rt + 8 * lhi + r;
        int col = d0 + 64 * wc + 16 * ct + lmod;
        float v = acc[rt][ct][r];
        out[(size_t)row * D_COLS + col] = 1.0f / (1.0f + __expf(-v));
      }
}

// ---------------------------------------------------------------------------
// Launch. Workspace: xb (16 MB) | xbt (16 MB) | inv_n (32 KB) = ~32.03 MB.
// ---------------------------------------------------------------------------
extern "C" void kernel_launch(void* const* d_in, const int* in_sizes, int n_in,
                              void* d_out, int out_size, void* d_ws,
                              size_t ws_size, hipStream_t stream) {
  (void)in_sizes; (void)n_in; (void)out_size; (void)ws_size;
  const float* x = (const float*)d_in[0];
  float* out = (float*)d_out;

  unsigned short* xb    = (unsigned short*)d_ws;
  unsigned short* xbt   = xb + (size_t)N_ROWS * D_COLS;
  float*          inv_n = (float*)(xbt + (size_t)N_ROWS * D_COLS);

  prep_norms_kernel<<<N_ROWS, 256, 0, stream>>>(x, xb, inv_n);
  dim3 tg(N_ROWS / 64, D_COLS / 64);
  transpose_bf16_kernel<<<tg, 256, 0, stream>>>(x, xbt);
  dim3 g(N_ROWS / 128, D_COLS / 128);
  cosattn_main_kernel<<<g, 256, 0, stream>>>(xb, xbt, inv_n, out);
}
